// SPC3_15298673508953
// MI455X (gfx1250) — compile-verified, matrix operand order checked by reading, not run on hardware
//
#include <hip/hip_runtime.h>
#include <hip/hip_bf16.h>
#include <math.h>

// Problem constants (from reference)
#define NBATCH 32
#define CCH    49     // channels == conv "depth" axis
#define HHW    81     // 9*9 spatial
#define MOUT   24     // conv output maps
#define KSZ    7
#define PADC   3
#define EPSV   1e-5f

// ---------------------------------------------------------------------------
// WMMA path selection: prefer full-precision f32 16x16x4 WMMA; fall back to
// the codegen-confirmed f16 16x16x32 WMMA (f32 accumulate) if unavailable.
// ---------------------------------------------------------------------------
#if __has_builtin(__builtin_amdgcn_wmma_f32_16x16x4_f32)
  #define USE_WMMA_F32 1
#else
  #define USE_WMMA_F32 0
#endif

typedef float    v2f  __attribute__((ext_vector_type(2)));
typedef float    v8f  __attribute__((ext_vector_type(8)));
typedef _Float16 v16h __attribute__((ext_vector_type(16)));

#if USE_WMMA_F32
  typedef float mat_t;
  #define KPAD 84      // 81 -> multiple of 4
#else
  typedef _Float16 mat_t;
  #define KPAD 96      // 81 -> multiple of 32
#endif

// leaky_relu(a*b) with b>=0  ==  g(a)*b,  g(z) = 0.505 z + 0.495 |z|
__device__ __forceinline__ float lrelu_fold(float z) {
    return 0.505f * z + 0.495f * fabsf(z);
}

// ---------------------------------------------------------------------------
// Kernel 0: zero the accumulator workspace
// ---------------------------------------------------------------------------
__global__ void spc3_zero(float* __restrict__ p, int nElems) {
    int i = blockIdx.x * blockDim.x + threadIdx.x;
    if (i < nElems) p[i] = 0.0f;
}

// ---------------------------------------------------------------------------
// Kernel 1: per (o, n) workgroup — conv + softmax + two WMMA GEMMs,
// atomically accumulates sum_d g(fk)*mask into acc[n,c,hw].
// 256 threads = 8 waves (wave32).
// ---------------------------------------------------------------------------
__global__ __launch_bounds__(256)
void spc3_fused(const float* __restrict__ x, const float* __restrict__ cw,
                const float* __restrict__ cb, float* __restrict__ acc) {
    __shared__ mat_t xs[64 * KPAD];   // x[n] tile   (rows c=0..48, cols hw, zero padded)
    __shared__ mat_t ms[64 * KPAD];   // mask tile   (rows d=0..48, cols hw, zero padded)
    __shared__ float fkG[64 * 64];    // g(fk)[d][c]

    const int o    = blockIdx.x;
    const int n    = blockIdx.y;
    const int tid  = threadIdx.x;
    const int lane = tid & 31;
    const int wave = tid >> 5;

    const float* __restrict__ xn = x + (size_t)n * CCH * HHW;

    // Prefetch the x tile into cache (lowers to global_prefetch_b8)
    if (tid < 64) __builtin_prefetch(xn + tid * 64, 0, 0);

    // Zero padded LDS tiles
    for (int i = tid; i < 64 * KPAD; i += 256) {
        xs[i] = (mat_t)0.0f;
        ms[i] = (mat_t)0.0f;
    }
    __syncthreads();

    // Stage x[n] (49x81) row-major into LDS
    for (int i = tid; i < CCH * HHW; i += 256) {
        int r = i / HHW, col = i - r * HHW;
        xs[r * KPAD + col] = (mat_t)xn[i];
    }

    // Conv weights for this output map (uniform -> scalar loads)
    float w[KSZ];
#pragma unroll
    for (int k = 0; k < KSZ; ++k) w[k] = cw[o * KSZ + k];
    const float bias = cb[o];
    __syncthreads();

    // -------- conv along channel axis + softmax over hw, one wave per row d
    for (int d = wave; d < CCH; d += 8) {
        float mv[3], ev[3];
        float mx = -3.0e38f;
#pragma unroll
        for (int j = 0; j < 3; ++j) {
            int col = lane + 32 * j;
            float a = bias;
            if (col < HHW) {
#pragma unroll
                for (int k = 0; k < KSZ; ++k) {
                    int rr = d + k - PADC;          // rr in [-3, 51]; rows >=49 are zero
                    if (rr >= 0) a += w[k] * (float)xs[rr * KPAD + col];
                }
                mx = fmaxf(mx, a);
            }
            mv[j] = a;
        }
#pragma unroll
        for (int s = 16; s > 0; s >>= 1) mx = fmaxf(mx, __shfl_xor(mx, s, 32));
        float sum = 0.0f;
#pragma unroll
        for (int j = 0; j < 3; ++j) {
            int col = lane + 32 * j;
            float e = (col < HHW) ? expf(mv[j] - mx) : 0.0f;
            ev[j] = e; sum += e;
        }
#pragma unroll
        for (int s = 16; s > 0; s >>= 1) sum += __shfl_xor(sum, s, 32);
        float inv = 1.0f / sum;
#pragma unroll
        for (int j = 0; j < 3; ++j) {
            int col = lane + 32 * j;
            if (col < HHW) ms[d * KPAD + col] = (mat_t)(ev[j] * inv);
        }
    }
    __syncthreads();

    const int col16 = lane & 15;           // N (and A-row M) index
    const int row8  = (lane >> 4) * 8;     // C/D fragment row offset
#if USE_WMMA_F32
    const int ko    = (lane >> 4) * 2;     // f32 A/B fragment K offset
#endif

    // -------- GEMM 1: fk[d,c] = mask(64xKPAD) @ x^T(KPADx64); store g(fk)
    // 16 output tiles (4x4), 2 per wave; all waves fully active (EXEC all 1s).
#pragma unroll
    for (int tt = 0; tt < 2; ++tt) {
        int t  = wave + tt * 8;
        int mt = (t >> 2) * 16;    // d tile base
        int nt = (t & 3) * 16;     // c tile base
        v8f c = {};
#if USE_WMMA_F32
        for (int kk = 0; kk < KPAD; kk += 4) {
            v2f a, b;
            a.x = (float)ms[(mt + col16) * KPAD + kk + ko];
            a.y = (float)ms[(mt + col16) * KPAD + kk + ko + 1];
            b.x = (float)xs[(nt + col16) * KPAD + kk + ko];      // B[k][n] = x[n][k]
            b.y = (float)xs[(nt + col16) * KPAD + kk + ko + 1];
            c = __builtin_amdgcn_wmma_f32_16x16x4_f32(false, a, false, b,
                                                      (short)0, c, false, false);
        }
#else
        for (int kk = 0; kk < KPAD; kk += 32) {
            v16h a, b;
#pragma unroll
            for (int v = 0; v < 8; ++v) {
                int kb = kk + ((v >= 4) ? 16 : 0) + (v & 3) * 2 + (lane >> 4) * 8;
                a[2 * v]     = ms[(mt + col16) * KPAD + kb];
                a[2 * v + 1] = ms[(mt + col16) * KPAD + kb + 1];
                b[2 * v]     = xs[(nt + col16) * KPAD + kb];
                b[2 * v + 1] = xs[(nt + col16) * KPAD + kb + 1];
            }
            c = __builtin_amdgcn_wmma_f32_16x16x32_f16(false, a, false, b,
                                                       (short)0, c, false, false);
        }
#endif
#pragma unroll
        for (int v = 0; v < 8; ++v) {
            fkG[(mt + v + row8) * 64 + (nt + col16)] = lrelu_fold(c[v]);
        }
    }
    __syncthreads();

    // -------- GEMM 2: out[c,hw] = g(fk)^T(64x64) @ mask(64xKPAD)
    // 24 output tiles (4 c-tiles x 6 hw-tiles), 3 per wave.
    for (int t = wave; t < 24; t += 8) {
        int mt = (t / 6) * 16;     // c tile base
        int nt = (t % 6) * 16;     // hw tile base
        v8f c = {};
        const int bc = nt + col16;
#if USE_WMMA_F32
        for (int kk = 0; kk < 64; kk += 4) {
            v2f a, b;
            a.x = fkG[(kk + ko) * 64 + mt + col16];       // A[m][k] = g(fk)[k][m]
            a.y = fkG[(kk + ko + 1) * 64 + mt + col16];
            b.x = (bc < KPAD) ? (float)ms[(kk + ko) * KPAD + bc] : 0.0f;
            b.y = (bc < KPAD) ? (float)ms[(kk + ko + 1) * KPAD + bc] : 0.0f;
            c = __builtin_amdgcn_wmma_f32_16x16x4_f32(false, a, false, b,
                                                      (short)0, c, false, false);
        }
#else
        for (int kk = 0; kk < 64; kk += 32) {
            v16h a, b;
#pragma unroll
            for (int v = 0; v < 8; ++v) {
                int kb = kk + ((v >= 4) ? 16 : 0) + (v & 3) * 2 + (lane >> 4) * 8;
                a[2 * v]     = (_Float16)fkG[kb * 64 + mt + col16];
                a[2 * v + 1] = (_Float16)fkG[(kb + 1) * 64 + mt + col16];
                b[2 * v]     = ms[kb * KPAD + bc];
                b[2 * v + 1] = ms[(kb + 1) * KPAD + bc];
            }
            c = __builtin_amdgcn_wmma_f32_16x16x32_f16(false, a, false, b,
                                                       (short)0, c, false, false);
        }
#endif
#pragma unroll
        for (int v = 0; v < 8; ++v) {
            int cc = mt + v + row8;
            int hw = nt + col16;
            if (cc < CCH && hw < HHW) {
                atomicAdd(&acc[((size_t)n * CCH + cc) * HHW + hw], c[v]);
            }
        }
    }
}

// ---------------------------------------------------------------------------
// Kernel 2: residual add + per-channel mean/var normalization (one block per c)
// ---------------------------------------------------------------------------
__global__ __launch_bounds__(256)
void spc3_norm(const float* __restrict__ x, const float* __restrict__ acc,
               const float* __restrict__ gamma, const float* __restrict__ beta,
               float* __restrict__ out) {
    const int c   = blockIdx.x;
    const int tid = threadIdx.x;
    __shared__ float red[16];

    float s = 0.0f, s2 = 0.0f;
    for (int i = tid; i < NBATCH * HHW; i += 256) {
        int nn = i / HHW, hw = i - nn * HHW;
        int idx = (nn * CCH + c) * HHW + hw;
        float v = acc[idx] + x[idx];
        s += v; s2 += v * v;
    }
#pragma unroll
    for (int m = 16; m > 0; m >>= 1) {
        s  += __shfl_xor(s,  m, 32);
        s2 += __shfl_xor(s2, m, 32);
    }
    if ((tid & 31) == 0) { red[tid >> 5] = s; red[8 + (tid >> 5)] = s2; }
    __syncthreads();
    if (tid == 0) {
        float ts = 0.0f, ts2 = 0.0f;
        for (int w = 0; w < 8; ++w) { ts += red[w]; ts2 += red[8 + w]; }
        const float invN = 1.0f / (float)(NBATCH * HHW);
        float mean = ts * invN;
        float var  = ts2 * invN - mean * mean;
        red[0] = mean;
        red[1] = rsqrtf(var + EPSV);
    }
    __syncthreads();
    const float mean = red[0], inv = red[1];
    const float g = gamma[c], b = beta[c];
    for (int i = tid; i < NBATCH * HHW; i += 256) {
        int nn = i / HHW, hw = i - nn * HHW;
        int idx = (nn * CCH + c) * HHW + hw;
        float v = acc[idx] + x[idx];
        out[idx] = (v - mean) * inv * g + b;
    }
}

// ---------------------------------------------------------------------------
extern "C" void kernel_launch(void* const* d_in, const int* in_sizes, int n_in,
                              void* d_out, int out_size, void* d_ws, size_t ws_size,
                              hipStream_t stream) {
    (void)in_sizes; (void)n_in; (void)out_size; (void)ws_size;
    const float* x     = (const float*)d_in[0];
    const float* cw    = (const float*)d_in[1];   // (24,1,7,1,1) -> [o*7+k]
    const float* cb    = (const float*)d_in[2];
    const float* gamma = (const float*)d_in[3];
    const float* beta  = (const float*)d_in[4];
    float* out = (float*)d_out;
    float* acc = (float*)d_ws;                    // NBATCH*CCH*HHW floats (~508 KB)

    const int total = NBATCH * CCH * HHW;
    spc3_zero<<<(total + 255) / 256, 256, 0, stream>>>(acc, total);

    dim3 g1(MOUT, NBATCH);                        // one WG per (o, n)
    spc3_fused<<<g1, 256, 0, stream>>>(x, cw, cb, acc);

    spc3_norm<<<CCH, 256, 0, stream>>>(x, acc, gamma, beta, out);
}
